// XConv_21483426415002
// MI455X (gfx1250) — compile-verified, matrix operand order checked by reading, not run on hardware
//
#include <hip/hip_runtime.h>

#define B_   8
#define N_   8192
#define M_   2048
#define K_   16
#define FIN_ 64
#define CPF_ 64
#define C_   128
#define DM_  4
#define CF_  128
#define KK_  256
#define GRP  32   // centers per workgroup (2 WMMA M-tiles)

typedef __attribute__((ext_vector_type(16))) _Float16 v16h;
typedef __attribute__((ext_vector_type(8)))  _Float16 v8h;
typedef __attribute__((ext_vector_type(8)))  float    v8f;

// LDS layout (dynamic shared, starts at LDS offset 0):
//   fa   : GRP*2048 f16              = 128 KB   [ci][k*128+c]
//   trs  : GRP*16*32 f16 (padded)    =  32 KB
//   tbuf : GRP*256 f32  (union: W2 cache 16KB / t-matrices 32KB / dw f16 32KB)
#define FA_HALVES   (GRP * 2048)
#define TRS_HALVES  (GRP * 512)
#define TBUF_LDS_OFF ((FA_HALVES + TRS_HALVES) * 2)   // byte offset of tbuf region

__device__ __forceinline__ float eluf(float x) { return x > 0.f ? x : (__expf(x) - 1.f); }

__device__ __forceinline__ v16h pack16(v8h lo, v8h hi) {
  v16h r;
#pragma unroll
  for (int e = 0; e < 8; ++e) { r[e] = lo[e]; r[8 + e] = hi[e]; }
  return r;
}

// 16-bit WMMA A/B^T fragment where the lane's 32 K-values are contiguous:
// VGPR0-3 hold K=8h..8h+7, VGPR4-7 hold K=16+8h..23+8h -> two 16B loads.
__device__ __forceinline__ v16h load_frag_contig(const _Float16* p) {
  v8h lo = *(const v8h*)p;
  v8h hi = *(const v8h*)(p + 16);
  return pack16(lo, hi);
}

// ---------------- KNN: one thread per (b, m), sorted top-16 -----------------
__global__ void knn_topk(const float* __restrict__ cxyz, const float* __restrict__ xyz,
                         int* __restrict__ idx) {
  int t = blockIdx.x * blockDim.x + threadIdx.x;
  if (t >= B_ * M_) return;
  int b = t / M_, m = t % M_;
  const float* cb = cxyz + (size_t)b * 3 * M_;
  float cx = cb[m], cy = cb[M_ + m], cz = cb[2 * M_ + m];
  const float* xb = xyz + (size_t)b * 3 * N_;
  float dist[K_]; int ind[K_];
#pragma unroll
  for (int i = 0; i < K_; ++i) { dist[i] = 3.4e38f; ind[i] = 0; }
  for (int n = 0; n < N_; ++n) {
    float dx = xb[n] - cx, dy = xb[N_ + n] - cy, dz = xb[2 * N_ + n] - cz;
    float d = dx * dx + dy * dy + dz * dz;
    if (d < dist[K_ - 1]) {
      int j = K_ - 1;
      while (j > 0 && dist[j - 1] > d) { dist[j] = dist[j - 1]; ind[j] = ind[j - 1]; --j; }
      dist[j] = d; ind[j] = n;
    }
  }
#pragma unroll
  for (int i = 0; i < K_; ++i) idx[(size_t)t * K_ + i] = ind[i];
}

// ------------- weight repacks to f16 (WMMA B^T row-major layouts) -----------
__global__ void repack_wt0(const float* __restrict__ wt0, _Float16* __restrict__ wt0h) {
  int i = blockIdx.x * blockDim.x + threadIdx.x;
  if (i >= KK_ * CF_ * K_) return;
  int o = i / (CF_ * K_);
  int kk = i % (CF_ * K_);
  int k = kk / CF_, c = kk % CF_;            // flat reduction index kk = k*128 + c
  wt0h[i] = (_Float16)wt0[(size_t)o * CF_ * K_ + c * K_ + k];
}

__global__ void repack_wpw(const float* __restrict__ wpw, _Float16* __restrict__ wpwh) {
  int i = blockIdx.x * blockDim.x + threadIdx.x;
  if (i >= C_ * CF_ * DM_) return;
  wpwh[i] = (_Float16)wpw[i];
}

// --------------------------- fused XConv main -------------------------------
// grid = B*M/32 workgroups, 256 threads (8 wave32), 32 centers per WG.
__global__ void __launch_bounds__(256)
xconv_main(const float* __restrict__ fts,
           const float* __restrict__ center_points,
           const float* __restrict__ points,
           const float* __restrict__ W1, const float* __restrict__ b1,
           const float* __restrict__ W2, const float* __restrict__ b2,
           const _Float16* __restrict__ Wt0h, const float* __restrict__ bt0,
           const float* __restrict__ Wt1, const float* __restrict__ bt1,
           const float* __restrict__ Wt2, const float* __restrict__ bt2,
           const float* __restrict__ Wdw,
           const _Float16* __restrict__ Wpwh, const float* __restrict__ bpw,
           const int* __restrict__ idx,
           float* __restrict__ out) {
  extern __shared__ char smem[];
  _Float16* fa    = (_Float16*)smem;                 // [32][2048]
  _Float16* trs   = fa + FA_HALVES;                  // [32][16][32] padded
  float*    tbuf  = (float*)(trs + TRS_HALVES);      // [32][256] f32 / W2$
  _Float16* dwbuf = (_Float16*)tbuf;                 // [32][512] f16 (union)

  const int tid = threadIdx.x;
  const int lane = tid & 31, wave = tid >> 5;
  const int r15 = lane & 15, hh = lane >> 4;
  const int grp = blockIdx.x;
  const int b = grp / (M_ / GRP);
  const int m0 = (grp % (M_ / GRP)) * GRP;

  // stage W2 (64x64 f32 = 16KB) into tbuf region via async global->LDS copies
  {
    for (int i = tid; i < (CPF_ * CPF_ * 4) / 16; i += 256) {
      unsigned voff = (unsigned)i * 16u;              // byte offset into W2
      unsigned ldso = (unsigned)TBUF_LDS_OFF + voff;  // LDS byte offset
      asm volatile("global_load_async_to_lds_b128 %0, %1, %2 offset:0"
                   :: "v"(ldso), "v"(voff), "s"(W2) : "memory");
    }
    asm volatile("s_wait_asynccnt 0x0" ::: "memory");
  }
  __syncthreads();

  // ---- step 1: gather + localize + mlp_d -> fa tiles (2 points/thread) ----
  for (int pt = tid; pt < GRP * K_; pt += 256) {
    const int ci = pt >> 4, kk = pt & 15;
    const int m = m0 + ci;
    const int n = idx[((size_t)(b * M_ + m)) * K_ + kk];
    const float* pb = points + (size_t)b * 3 * N_;
    const float* cp = center_points + (size_t)b * 3 * M_;
    float l0 = pb[n] - cp[m];
    float l1 = pb[N_ + n] - cp[M_ + m];
    float l2 = pb[2 * N_ + n] - cp[2 * M_ + m];
    float h[CPF_];
#pragma unroll 8
    for (int o = 0; o < CPF_; ++o)
      h[o] = eluf(W1[o * 3] * l0 + W1[o * 3 + 1] * l1 + W1[o * 3 + 2] * l2 + b1[o]);
    _Float16* dst = fa + ci * 2048 + kk * CF_;
    for (int o = 0; o < CPF_; ++o) {
      float acc = b2[o];
#pragma unroll 8
      for (int i2 = 0; i2 < CPF_; ++i2) acc += tbuf[o * CPF_ + i2] * h[i2];
      dst[o] = (_Float16)eluf(acc);
    }
    const float* fb = fts + (size_t)b * FIN_ * N_;
    for (int c = 0; c < FIN_; ++c) dst[CPF_ + c] = (_Float16)fb[(size_t)c * N_ + n];
  }
  __syncthreads();

  // ---- step 2: GEMM1  t0[32,256] = fa[32,2048] x Wt0h^T -------------------
  // Per wave: 2 M-tiles x 2 N-tiles in one k-loop -> 4 independent WMMA
  // chains; each A and B fragment feeds two WMMAs.
  {
    const int nt0 = wave * 2, nt1 = wave * 2 + 1;
    v8f a00 = {0.f, 0.f, 0.f, 0.f, 0.f, 0.f, 0.f, 0.f};
    v8f a01 = a00, a10 = a00, a11 = a00;
    const _Float16* arow0 = fa + r15 * 2048 + 8 * hh;          // centers 0..15
    const _Float16* arow1 = arow0 + 16 * 2048;                 // centers 16..31
    const _Float16* brow0 = Wt0h + (size_t)(nt0 * 16 + r15) * 2048 + 8 * hh;
    const _Float16* brow1 = brow0 + 16 * 2048;
    for (int ks = 0; ks < 64; ++ks) {
      v16h bf0 = load_frag_contig(brow0 + ks * 32);
      v16h bf1 = load_frag_contig(brow1 + ks * 32);
      v16h af0 = load_frag_contig(arow0 + ks * 32);
      v16h af1 = load_frag_contig(arow1 + ks * 32);
      a00 = __builtin_amdgcn_wmma_f32_16x16x32_f16(false, af0, false, bf0,
                                                   (short)0, a00, false, false);
      a10 = __builtin_amdgcn_wmma_f32_16x16x32_f16(false, af1, false, bf0,
                                                   (short)0, a10, false, false);
      a01 = __builtin_amdgcn_wmma_f32_16x16x32_f16(false, af0, false, bf1,
                                                   (short)0, a01, false, false);
      a11 = __builtin_amdgcn_wmma_f32_16x16x32_f16(false, af1, false, bf1,
                                                   (short)0, a11, false, false);
    }
    const int o0 = nt0 * 16 + r15;
    const int o1 = nt1 * 16 + r15;
    const float bo0 = bt0[o0], bo1 = bt0[o1];
#pragma unroll
    for (int r = 0; r < 8; ++r) {
      int ci = r + 8 * hh;
      tbuf[ci * 256 + o0]        = eluf(a00[r] + bo0);
      tbuf[(16 + ci) * 256 + o0] = eluf(a10[r] + bo0);
      tbuf[ci * 256 + o1]        = eluf(a01[r] + bo1);
      tbuf[(16 + ci) * 256 + o1] = eluf(a11[r] + bo1);
    }
  }
  __syncthreads();

  // ---- step 3: per-row Wt1/Wt2 convs -> trans (f16, rows padded to 32) ----
  for (int rr = tid; rr < GRP * 16; rr += 256) {
    const int ci = rr >> 4, c = rr & 15;
    float row[K_], t1[K_];
#pragma unroll
    for (int k = 0; k < K_; ++k) row[k] = tbuf[ci * 256 + c * 16 + k];
#pragma unroll
    for (int d = 0; d < K_; ++d) {
      float a = bt1[c * 16 + d];
#pragma unroll
      for (int k = 0; k < K_; ++k) a += Wt1[c * 256 + d * 16 + k] * row[k];
      t1[d] = eluf(a);
    }
    _Float16* tr = trs + ci * 512 + c * 32;
#pragma unroll
    for (int d = 0; d < K_; ++d) {
      float a = bt2[c * 16 + d];
#pragma unroll
      for (int k = 0; k < K_; ++k) a += Wt2[c * 256 + d * 16 + k] * t1[k];
      tr[d] = (_Float16)a;
      tr[16 + d] = (_Float16)0.f;   // zero K-pad
    }
  }
  __syncthreads();

  // ---- step 4: fx = trans(16x16,pad32) @ fa(16x128) via WMMA, fuse Wdw ----
  // Two adjacent c-tiles per iteration: same center -> one A fragment feeds
  // two WMMAs into independent accumulators (fills hazard slots).
  {
    for (int jj = 0; jj < 16; ++jj) {
      const int jid = wave * 32 + 2 * jj;
      const int ci = jid >> 3;
      const int ct0 = jid & 7, ct1 = ct0 + 1;
      v16h af = load_frag_contig(trs + ci * 512 + r15 * 32 + 8 * hh); // A row = i
      v16h bf0, bf1;
#pragma unroll
      for (int e = 0; e < 16; ++e) {
        int p = e >> 1;
        int kb = (p < 4) ? 2 * p : (16 + 2 * (p - 4));
        int k = kb + 8 * hh + (e & 1);
        bool valid = (k < K_);
        const _Float16* src = fa + ci * 2048 + k * CF_ + r15;
        bf0[e] = valid ? src[ct0 * 16] : (_Float16)0.f;
        bf1[e] = valid ? src[ct1 * 16] : (_Float16)0.f;
      }
      v8f accz = {0.f, 0.f, 0.f, 0.f, 0.f, 0.f, 0.f, 0.f};
      v8f acc0 = __builtin_amdgcn_wmma_f32_16x16x32_f16(false, af, false, bf0,
                                                        (short)0, accz, false, false);
      v8f acc1 = __builtin_amdgcn_wmma_f32_16x16x32_f16(false, af, false, bf1,
                                                        (short)0, accz, false, false);
      // depthwise Wdw on the accumulator fragments: rows = neighbor k, col = c
      const int cc0 = ct0 * 16 + r15;
      const int cc1 = ct1 * 16 + r15;
      float d00 = 0.f, d01 = 0.f, d02 = 0.f, d03 = 0.f;
      float d10 = 0.f, d11 = 0.f, d12 = 0.f, d13 = 0.f;
#pragma unroll
      for (int r = 0; r < 8; ++r) {
        int krow = r + 8 * hh;
        float fx0 = acc0[r];
        float fx1 = acc1[r];
        d00 += Wdw[cc0 * 64 + 0 * 16 + krow] * fx0;
        d01 += Wdw[cc0 * 64 + 1 * 16 + krow] * fx0;
        d02 += Wdw[cc0 * 64 + 2 * 16 + krow] * fx0;
        d03 += Wdw[cc0 * 64 + 3 * 16 + krow] * fx0;
        d10 += Wdw[cc1 * 64 + 0 * 16 + krow] * fx1;
        d11 += Wdw[cc1 * 64 + 1 * 16 + krow] * fx1;
        d12 += Wdw[cc1 * 64 + 2 * 16 + krow] * fx1;
        d13 += Wdw[cc1 * 64 + 3 * 16 + krow] * fx1;
      }
      d00 += __shfl_xor(d00, 16);  d01 += __shfl_xor(d01, 16);
      d02 += __shfl_xor(d02, 16);  d03 += __shfl_xor(d03, 16);
      d10 += __shfl_xor(d10, 16);  d11 += __shfl_xor(d11, 16);
      d12 += __shfl_xor(d12, 16);  d13 += __shfl_xor(d13, 16);
      if (hh == 0) {
        dwbuf[ci * 512 + cc0 * 4 + 0] = (_Float16)d00;
        dwbuf[ci * 512 + cc0 * 4 + 1] = (_Float16)d01;
        dwbuf[ci * 512 + cc0 * 4 + 2] = (_Float16)d02;
        dwbuf[ci * 512 + cc0 * 4 + 3] = (_Float16)d03;
        dwbuf[ci * 512 + cc1 * 4 + 0] = (_Float16)d10;
        dwbuf[ci * 512 + cc1 * 4 + 1] = (_Float16)d11;
        dwbuf[ci * 512 + cc1 * 4 + 2] = (_Float16)d12;
        dwbuf[ci * 512 + cc1 * 4 + 3] = (_Float16)d13;
      }
    }
  }
  __syncthreads();

  // ---- step 5: GEMM2  out[32,128] = dw[32,512] x Wpw^T, bias + ReLU -------
  {
    v8f acc0 = {0.f, 0.f, 0.f, 0.f, 0.f, 0.f, 0.f, 0.f};
    v8f acc1 = acc0;
    const _Float16* arow0 = dwbuf + r15 * 512 + 8 * hh;
    const _Float16* arow1 = arow0 + 16 * 512;
    const _Float16* brow = Wpwh + (size_t)(wave * 16 + r15) * 512 + 8 * hh;
    for (int ks = 0; ks < 16; ++ks) {
      v16h bf  = load_frag_contig(brow + ks * 32);
      v16h af0 = load_frag_contig(arow0 + ks * 32);
      v16h af1 = load_frag_contig(arow1 + ks * 32);
      acc0 = __builtin_amdgcn_wmma_f32_16x16x32_f16(false, af0, false, bf,
                                                    (short)0, acc0, false, false);
      acc1 = __builtin_amdgcn_wmma_f32_16x16x32_f16(false, af1, false, bf,
                                                    (short)0, acc1, false, false);
    }
    const int o = wave * 16 + r15;
    const float bo = bpw[o];
#pragma unroll
    for (int r = 0; r < 8; ++r) {
      int ci = r + 8 * hh;
      float v0 = acc0[r] + bo;
      float v1 = acc1[r] + bo;
      out[(size_t)b * C_ * M_ + (size_t)o * M_ + (m0 + ci)]      = v0 > 0.f ? v0 : 0.f;
      out[(size_t)b * C_ * M_ + (size_t)o * M_ + (m0 + 16 + ci)] = v1 > 0.f ? v1 : 0.f;
    }
  }
}

extern "C" void kernel_launch(void* const* d_in, const int* in_sizes, int n_in,
                              void* d_out, int out_size, void* d_ws, size_t ws_size,
                              hipStream_t stream) {
  const float* fts  = (const float*)d_in[0];
  const float* cxyz = (const float*)d_in[1];
  const float* cpts = (const float*)d_in[2];
  const float* xyz  = (const float*)d_in[3];
  const float* pts  = (const float*)d_in[4];
  const float* W1   = (const float*)d_in[5];
  const float* b1   = (const float*)d_in[6];
  const float* W2   = (const float*)d_in[7];
  const float* b2   = (const float*)d_in[8];
  const float* Wt0  = (const float*)d_in[9];
  const float* bt0  = (const float*)d_in[10];
  const float* Wt1  = (const float*)d_in[11];
  const float* bt1  = (const float*)d_in[12];
  const float* Wt2  = (const float*)d_in[13];
  const float* bt2  = (const float*)d_in[14];
  const float* Wdw  = (const float*)d_in[15];
  const float* Wpw  = (const float*)d_in[16];
  const float* bpw  = (const float*)d_in[17];
  float* out = (float*)d_out;

  char* ws = (char*)d_ws;
  int* idx       = (int*)ws;                       // B*M*K ints   = 1 MB
  _Float16* wt0h = (_Float16*)(ws + (1 << 20));    // 256*2048 f16 = 1 MB
  _Float16* wpwh = (_Float16*)(ws + (2 << 20));    // 128*512 f16  = 128 KB

  knn_topk<<<(B_ * M_ + 255) / 256, 256, 0, stream>>>(cxyz, xyz, idx);
  repack_wt0<<<(KK_ * CF_ * K_ + 255) / 256, 256, 0, stream>>>(Wt0, wt0h);
  repack_wpw<<<(C_ * CF_ * DM_ + 255) / 256, 256, 0, stream>>>(Wpw, wpwh);

  size_t shm = (size_t)(FA_HALVES + TRS_HALVES) * sizeof(_Float16)
             + (size_t)GRP * 256 * sizeof(float);   // 192 KB
  xconv_main<<<(B_ * M_) / GRP, 256, shm, stream>>>(
      fts, cpts, pts, W1, b1, W2, b2, wt0h, bt0, Wt1, bt1, Wt2, bt2,
      Wdw, wpwh, bpw, idx, out);
}
